// TemporalTransformerHawkesGraphModel_40269613367671
// MI455X (gfx1250) — matrix operand order repaired and checked by reading, not compiled, not run on hardware
//
#include <hip/hip_runtime.h>
#include <hip/hip_bf16.h>

typedef __attribute__((ext_vector_type(16))) __bf16 v16bf;
typedef __attribute__((ext_vector_type(8)))  float  v8f;

#define N_ENT_C   40000
#define D_C       200
#define BS_C      512
#define HIST_C    8
#define TOTAL_C   262144
#define KP        224      // padded K (7 * 32)
#define NP        208      // padded N (13 * 16)
#define NTILES_D  13
#define KTILES    7
#define NLOC      512

union FragU { v16bf v; uint4 u[2]; };

// A-matrix 16x32 bf16 fragment: lane holds row M=lane&15; VGPR0-3 = K kbase..kbase+7,
// VGPR4-7 = K kbase+16..kbase+23, kbase = 8*(lane>>4). Two 16B chunks, 16 elems apart.
__device__ __forceinline__ v16bf load_frag_a(const unsigned short* row, int k0) {
  FragU f;
  f.u[0] = *reinterpret_cast<const uint4*>(row + k0);
  f.u[1] = *reinterpret_cast<const uint4*>(row + k0 + 16);
  return f.v;
}
// B-matrix 32x16 bf16 fragment: lane holds col N=lane&15; same K layout as A:
// VGPR0-3 = K kbase..kbase+7, VGPR4-7 = K kbase+16..kbase+23, kbase = 8*(lane>>4).
__device__ __forceinline__ v16bf load_frag_b(const unsigned short* col, int k0) {
  FragU f;
  f.u[0] = *reinterpret_cast<const uint4*>(col + k0);
  f.u[1] = *reinterpret_cast<const uint4*>(col + k0 + 16);
  return f.v;
}

__device__ __forceinline__ unsigned short f2bf(float x) {
  unsigned int u = __float_as_uint(x);
  u += 0x7FFFu + ((u >> 16) & 1u);     // round-to-nearest-even
  return (unsigned short)(u >> 16);
}

// softplus(10x)/10, numerically stable
__device__ __forceinline__ float softplus10(float x) {
  return fmaxf(x, 0.0f) + 0.1f * log1pf(expf(-10.0f * fabsf(x)));
}

// ---------------- prep: ent_embeds f32[40000][200] -> bf16[40000][224] (zero pad) ----
__global__ void k_prep_ent(const float* __restrict__ ent, unsigned short* __restrict__ out) {
  int idx = blockIdx.x * blockDim.x + threadIdx.x;
  if (idx >= N_ENT_C * KP) return;
  int e = idx / KP, k = idx - e * KP;
  float v = (k < D_C) ? ent[(size_t)e * D_C + k] : 0.0f;
  out[idx] = f2bf(v);
}

// ---------------- prep: Wg f32[200][200] -> WgT bf16[208][224] column-per-row --------
__global__ void k_prep_wgt(const float* __restrict__ Wg, unsigned short* __restrict__ out) {
  int idx = blockIdx.x * blockDim.x + threadIdx.x;
  if (idx >= NP * KP) return;
  int n = idx / KP, k = idx - n * KP;
  float v = (n < D_C && k < D_C) ? Wg[(size_t)k * D_C + n] : 0.0f;
  out[idx] = f2bf(v);
}

// ------- ghq: gh rows for the 4096 (b,h) query positions, via WMMA ------------------
// ghq[q][col] = relu( ent_bf16[node_ids[bni[q]]] @ WgT )   q = b*8+h
__global__ void k_ghq(const unsigned short* __restrict__ entb,
                      const unsigned short* __restrict__ wgt,
                      const int* __restrict__ node_ids,
                      const int* __restrict__ bni,
                      float* __restrict__ ghq) {
  int wid  = (blockIdx.x * blockDim.x + threadIdx.x) >> 5;
  int lane = threadIdx.x & 31;
  int mtile = wid / NTILES_D;
  int ntile = wid - mtile * NTILES_D;
  int r = lane & 15, hi = lane >> 4;
  int q = mtile * 16 + r;                       // < 4096
  int node = node_ids[bni[q]];
  const unsigned short* arow = entb + (size_t)node * KP;
  const unsigned short* bcol = wgt + (size_t)(ntile * 16 + r) * KP;
  int ka = hi * 8, kb = hi * 8;
  v8f c = {};
#pragma unroll
  for (int kt = 0; kt < KTILES; ++kt) {
    v16bf a = load_frag_a(arow, kt * 32 + ka);
    v16bf b = load_frag_b(bcol, kt * 32 + kb);
    c = __builtin_amdgcn_wmma_f32_16x16x32_bf16(false, a, false, b, (short)0, c, false, false);
  }
  int col = ntile * 16 + r;
#pragma unroll
  for (int j = 0; j < 8; ++j) {
    int row = mtile * 16 + j + 8 * hi;
    ghq[(size_t)row * NP + col] = fmaxf(c[j], 0.0f);
  }
}

// ------- inten_raw: concat(qe, masked-mean(ghq), qr) @ W_inten  --------------------
__global__ void k_inten(const float* __restrict__ ent, const float* __restrict__ rel,
                        const float* __restrict__ W_inten, const float* __restrict__ hist_t,
                        const int* __restrict__ q_ent, const int* __restrict__ q_rel,
                        const float* __restrict__ ghq,
                        float* __restrict__ inten_f, unsigned short* __restrict__ inten_b) {
  __shared__ float cat[3 * D_C];
  int b = blockIdx.x, tid = threadIdx.x;
  int qe = q_ent[b], qr = q_rel[b];
  if (tid < D_C) {
    cat[tid] = ent[(size_t)qe * D_C + tid];
    float s = 0.0f, cnt = 0.0f;
#pragma unroll
    for (int h = 0; h < HIST_C; ++h) {
      if (hist_t[b * HIST_C + h] != -1.0f) {
        s += ghq[(size_t)(b * HIST_C + h) * NP + tid];
        cnt += 1.0f;
      }
    }
    cat[D_C + tid] = s / fmaxf(cnt, 1.0f);
    cat[2 * D_C + tid] = rel[(size_t)qr * D_C + tid];
  }
  __syncthreads();
  if (tid < D_C) {
    float acc = 0.0f;
    for (int i = 0; i < 3 * D_C; ++i)
      acc += cat[i] * W_inten[(size_t)i * D_C + tid];
    inten_f[b * NP + tid] = acc;
    inten_b[b * KP + tid] = f2bf(acc);
  } else {
    if (tid < NP) inten_f[b * NP + tid] = 0.0f;
    if (tid < KP) inten_b[b * KP + tid] = 0;
  }
}

// ------- global intensities: out[b][e] = softplus10( inten_bf16[b] . ent_bf16[e] ) --
__global__ void k_global(const unsigned short* __restrict__ intenb,
                         const unsigned short* __restrict__ entb,
                         float* __restrict__ out) {
  int wid  = (blockIdx.x * blockDim.x + threadIdx.x) >> 5;
  int lane = threadIdx.x & 31;
  int mtile = wid & 31;         // 512/16 = 32 row tiles
  int ntile = wid >> 5;         // 40000/16 = 2500 col tiles
  int r = lane & 15, hi = lane >> 4;
  const unsigned short* arow = intenb + (size_t)(mtile * 16 + r) * KP;
  const unsigned short* bcol = entb   + (size_t)(ntile * 16 + r) * KP;
  int ka = hi * 8, kb = hi * 8;
  v16bf a[KTILES];
#pragma unroll
  for (int kt = 0; kt < KTILES; ++kt) a[kt] = load_frag_a(arow, kt * 32 + ka);
  v8f c = {};
#pragma unroll
  for (int kt = 0; kt < KTILES; ++kt) {
    v16bf bfr = load_frag_b(bcol, kt * 32 + kb);
    c = __builtin_amdgcn_wmma_f32_16x16x32_bf16(false, a[kt], false, bfr, (short)0, c, false, false);
  }
  int e = ntile * 16 + r;
#pragma unroll
  for (int j = 0; j < 8; ++j) {
    int row = mtile * 16 + j + 8 * hi;
    out[(size_t)row * N_ENT_C + e] = softplus10(c[j]);
  }
}

// ------- fused local intensities: never materializes gh -----------------------------
// lint[t] = softplus10( sum_col relu((ent[node_ids[t]] @ WgT)[col]) * inten_f[b][col] )
__global__ void k_local(const unsigned short* __restrict__ entb,
                        const unsigned short* __restrict__ wgt,
                        const int* __restrict__ node_ids,
                        const float* __restrict__ inten_f,
                        float* __restrict__ lint) {
  int wid  = (blockIdx.x * blockDim.x + threadIdx.x) >> 5;   // 16384 row tiles
  int lane = threadIdx.x & 31;
  int r = lane & 15, hi = lane >> 4;
  int t = wid * 16 + r;
  int node = node_ids[t];
  const unsigned short* arow = entb + (size_t)node * KP;
  int ka = hi * 8, kb = hi * 8;
  v16bf a[KTILES];
#pragma unroll
  for (int kt = 0; kt < KTILES; ++kt) a[kt] = load_frag_a(arow, kt * 32 + ka);
  int b = wid >> 5;                                  // (wid*16)/512
  float lacc[8];
#pragma unroll
  for (int j = 0; j < 8; ++j) lacc[j] = 0.0f;
  for (int ntile = 0; ntile < NTILES_D; ++ntile) {
    const unsigned short* bcol = wgt + (size_t)(ntile * 16 + r) * KP;
    v8f c = {};
#pragma unroll
    for (int kt = 0; kt < KTILES; ++kt) {
      v16bf bfr = load_frag_b(bcol, kt * 32 + kb);
      c = __builtin_amdgcn_wmma_f32_16x16x32_bf16(false, a[kt], false, bfr, (short)0, c, false, false);
    }
    float w = inten_f[b * NP + ntile * 16 + r];      // pad cols are zero
#pragma unroll
    for (int j = 0; j < 8; ++j) lacc[j] += fmaxf(c[j], 0.0f) * w;
  }
  // reduce across the 16 columns held by lanes [g*16 .. g*16+15]
#pragma unroll
  for (int j = 0; j < 8; ++j) {
    float v = lacc[j];
    v += __shfl_xor(v, 1, 32);
    v += __shfl_xor(v, 2, 32);
    v += __shfl_xor(v, 4, 32);
    v += __shfl_xor(v, 8, 32);
    lacc[j] = v;
  }
  if (r == 0) {
#pragma unroll
    for (int j = 0; j < 8; ++j) {
      int row = wid * 16 + j + 8 * hi;               // flat node index
      lint[row] = softplus10(lacc[j]);
    }
  }
}

// ------- scatter-mean fixup: out[b][e] = (sp_g + sum sp_local) / (1 + cnt) ----------
__global__ void k_scatter(const int* __restrict__ node_ids,
                          const float* __restrict__ lint,
                          float* __restrict__ out) {
  __shared__ int   se[NLOC];
  __shared__ float sv[NLOC];
  int b = blockIdx.x, n = threadIdx.x;
  int t = b * NLOC + n;
  se[n] = node_ids[t];
  sv[n] = lint[t];
  __syncthreads();
  int e = se[n];
  int first = n;
  float sum = 0.0f, cnt = 0.0f;
  for (int m = 0; m < NLOC; ++m) {
    if (se[m] == e) {
      sum += sv[m];
      cnt += 1.0f;
      if (m < first) first = m;
    }
  }
  if (first == n) {
    size_t idx = (size_t)b * N_ENT_C + e;
    out[idx] = (out[idx] + sum) / (1.0f + cnt);
  }
}

extern "C" void kernel_launch(void* const* d_in, const int* in_sizes, int n_in,
                              void* d_out, int out_size, void* d_ws, size_t ws_size,
                              hipStream_t stream) {
  const float* ent     = (const float*)d_in[0];
  const float* rel     = (const float*)d_in[1];
  const float* Wg      = (const float*)d_in[2];
  const float* W_inten = (const float*)d_in[3];
  const float* hist_t  = (const float*)d_in[4];
  // d_in[5] query_timestamps: unused by the reference
  const int* q_ent     = (const int*)d_in[6];
  const int* q_rel     = (const int*)d_in[7];
  const int* node_ids  = (const int*)d_in[8];
  const int* bni       = (const int*)d_in[9];
  float* out = (float*)d_out;

  char* ws = (char*)d_ws;
  size_t off = 0;
  auto carve = [&](size_t bytes) {
    char* p = ws + off;
    off = (off + bytes + 255) & ~(size_t)255;
    return p;
  };
  unsigned short* entb   = (unsigned short*)carve((size_t)N_ENT_C * KP * 2); // 17.9 MB
  unsigned short* wgt    = (unsigned short*)carve((size_t)NP * KP * 2);      // 93 KB
  float*          ghq    = (float*)         carve((size_t)BS_C * HIST_C * NP * 4); // 3.4 MB
  float*          intenf = (float*)         carve((size_t)BS_C * NP * 4);
  unsigned short* intenb = (unsigned short*)carve((size_t)BS_C * KP * 2);
  float*          lint   = (float*)         carve((size_t)TOTAL_C * 4);      // 1 MB
  (void)ws_size; (void)in_sizes; (void)n_in; (void)out_size;

  k_prep_ent<<<(N_ENT_C * KP + 255) / 256, 256, 0, stream>>>(ent, entb);
  k_prep_wgt<<<(NP * KP + 255) / 256, 256, 0, stream>>>(Wg, wgt);
  // 4096/16 = 256 row tiles * 13 col tiles = 3328 waves -> 416 blocks of 8 waves
  k_ghq<<<416, 256, 0, stream>>>(entb, wgt, node_ids, bni, ghq);
  k_inten<<<BS_C, 256, 0, stream>>>(ent, rel, W_inten, hist_t, q_ent, q_rel,
                                    ghq, intenf, intenb);
  // 32 * 2500 = 80000 waves -> 10000 blocks
  k_global<<<10000, 256, 0, stream>>>(intenb, entb, out);
  // 262144/16 = 16384 waves -> 2048 blocks
  k_local<<<2048, 256, 0, stream>>>(entb, wgt, node_ids, intenf, lint);
  k_scatter<<<BS_C, NLOC, 0, stream>>>(node_ids, lint, out);
}
